// MultiHeadAttention_38087769980988
// MI455X (gfx1250) — matrix operand order repaired and checked by reading, not compiled, not run on hardware
//
#include <hip/hip_runtime.h>
#include <hip/hip_bf16.h>

#define Bd 2
#define Ld 2048
#define Dd 1024
#define Hd 16
#define DKd 64
#define BLd (Bd * Ld)

typedef __attribute__((ext_vector_type(16))) __bf16 v16bf;
typedef __attribute__((ext_vector_type(8)))  __bf16 v8bf;
typedef __attribute__((ext_vector_type(8)))  float  v8f;

// ---------------------------------------------------------------------------
// WMMA helper: D = A(16x32 bf16) * B(32x16 bf16) + C(16x16 f32)
// ---------------------------------------------------------------------------
__device__ __forceinline__ v8f wmma_bf16(v16bf a, v16bf b, v8f c) {
    return __builtin_amdgcn_wmma_f32_16x16x32_bf16(
        /*neg_a=*/false, a, /*neg_b=*/false, b,
        /*c_mod=*/(short)0, c, /*reuse_a=*/false, /*reuse_b=*/false);
}

// A-fragment (16x32 bf16, row-major source). Lane holds row (lane&15).
// half = lane>>4. Element e in 0..15 -> K = 16*half + e
__device__ __forceinline__ v16bf load_a_frag(const __bf16* rowptr, int half) {
    const v8bf lo = *(const v8bf*)(rowptr + 16 * half);
    const v8bf hi = *(const v8bf*)(rowptr + 16 * half + 8);
    v16bf r;
#pragma unroll
    for (int i = 0; i < 8; ++i) { r[i] = lo[i]; r[i + 8] = hi[i]; }
    return r;
}

// B-fragment (32x16 bf16): lane = column, element e -> K = 16*half + e.
// Caller arranges a contiguous 16-element (32B, aligned) run per lane.
__device__ __forceinline__ v16bf load_b_frag(const __bf16* p) {
    return *(const v16bf*)p;
}

// ---------------------------------------------------------------------------
// fp32 -> bf16 conversion (grid-stride)
// ---------------------------------------------------------------------------
__global__ void cvt_f32_bf16_kernel(const float* __restrict__ in,
                                    __bf16* __restrict__ out, size_t n) {
    size_t i = (size_t)blockIdx.x * blockDim.x + threadIdx.x;
    size_t stride = (size_t)gridDim.x * blockDim.x;
    for (; i < n; i += stride) out[i] = (__bf16)in[i];
}

// W[k][n] fp32 -> Wt[n][k] bf16  (D x D)
__global__ void transpose_w_kernel(const float* __restrict__ W,
                                   __bf16* __restrict__ Wt, int d) {
    size_t n = (size_t)d * d;
    size_t i = (size_t)blockIdx.x * blockDim.x + threadIdx.x;
    size_t stride = (size_t)gridDim.x * blockDim.x;
    for (; i < n; i += stride) {
        int k = (int)(i / d);
        int c = (int)(i % d);
        Wt[(size_t)c * d + k] = (__bf16)W[i];
    }
}

// ---------------------------------------------------------------------------
// GEMM: out = A[M][K] * Bt[N][K]^T + bias.  One wave -> 16x64 output tile.
// Software-pipelined: next k-step's 5 fragments are loaded while the 4 WMMAs
// of the current k-step execute back-to-back.
// mode 0: bf16 row-major out [M][N]
// mode 1: bf16 "V transposed" out: Vt[b][h][dk][L]  (M = B*L, N = H*DK)
// mode 2: fp32 row-major out [M][N]
// ---------------------------------------------------------------------------
__global__ __launch_bounds__(256) void gemm_bf16_kernel(
    const __bf16* __restrict__ A, const __bf16* __restrict__ Bt,
    const float* __restrict__ bias, void* __restrict__ out,
    int M, int N, int K, int mode) {
    const int wave = threadIdx.x >> 5;
    const int lane = threadIdx.x & 31;
    const int half = lane >> 4;
    const int ln   = lane & 15;
    const int gw = blockIdx.x * (blockDim.x >> 5) + wave;
    const int colGroups = N / 64;
    const int rt = gw / colGroups;
    const int cg = gw % colGroups;
    if (rt * 16 >= M) return;
    const int row0 = rt * 16;
    const int col0 = cg * 64;

    const __bf16* arow = A + (size_t)(row0 + ln) * K;
    const __bf16* br0 = Bt + (size_t)(col0 + 0 * 16 + ln) * K + 16 * half;
    const __bf16* br1 = Bt + (size_t)(col0 + 1 * 16 + ln) * K + 16 * half;
    const __bf16* br2 = Bt + (size_t)(col0 + 2 * 16 + ln) * K + 16 * half;
    const __bf16* br3 = Bt + (size_t)(col0 + 3 * 16 + ln) * K + 16 * half;

    v8f acc[4] = { {}, {}, {}, {} };

    // Prologue: fragments for k = 0.
    v16bf af = load_a_frag(arow, half);
    v16bf b0 = load_b_frag(br0);
    v16bf b1 = load_b_frag(br1);
    v16bf b2 = load_b_frag(br2);
    v16bf b3 = load_b_frag(br3);

    for (int k = 0; k < K; k += 32) {
        // Issue next iteration's loads first (clamped on the last step so the
        // body stays branch-free and in-bounds; redundant final reload is cheap).
        const int kn = (k + 32 < K) ? (k + 32) : k;
        __builtin_prefetch(arow + kn + 64, 0, 1);
        v16bf afn = load_a_frag(arow + kn, half);
        v16bf b0n = load_b_frag(br0 + kn);
        v16bf b1n = load_b_frag(br1 + kn);
        v16bf b2n = load_b_frag(br2 + kn);
        v16bf b3n = load_b_frag(br3 + kn);

        // 4 back-to-back WMMAs on the previously loaded fragments.
        acc[0] = wmma_bf16(af, b0, acc[0]);
        acc[1] = wmma_bf16(af, b1, acc[1]);
        acc[2] = wmma_bf16(af, b2, acc[2]);
        acc[3] = wmma_bf16(af, b3, acc[3]);

        af = afn; b0 = b0n; b1 = b1n; b2 = b2n; b3 = b3n;
    }

#pragma unroll
    for (int t = 0; t < 4; ++t) {
        const int col = col0 + t * 16 + ln;
        const float bv = bias ? bias[col] : 0.0f;
#pragma unroll
        for (int r = 0; r < 8; ++r) {
            const int m = row0 + r + 8 * half;
            const float val = acc[t][r] + bv;
            if (mode == 0) {
                ((__bf16*)out)[(size_t)m * N + col] = (__bf16)val;
            } else if (mode == 1) {
                const int b = m / Ld, l = m % Ld;
                const int h = col / DKd, dk = col % DKd;
                ((__bf16*)out)[(((size_t)(b * Hd + h)) * DKd + dk) * Ld + l] = (__bf16)val;
            } else {
                ((float*)out)[(size_t)m * N + col] = val;
            }
        }
    }
}

// ---------------------------------------------------------------------------
// Flash attention: 1 wave per (b, h, 16-query tile), 32-key blocks,
// online softmax, causal mask. Q,K: bf16 [B*L][D]; Vt: bf16 [B*H][DK][L].
// AO: bf16 [B*L][D].  K/V fragments are double-buffered: the next block's
// 8 loads issue right after the current score WMMAs and complete under the
// softmax VALU/LDS work.
// ---------------------------------------------------------------------------
__global__ __launch_bounds__(256) void flash_attn_kernel(
    const __bf16* __restrict__ Q, const __bf16* __restrict__ Km,
    const __bf16* __restrict__ Vt, __bf16* __restrict__ AO) {
    const int wave = threadIdx.x >> 5;
    const int lane = threadIdx.x & 31;
    const int half = lane >> 4;
    const int ln   = lane & 15;
    const int gw = blockIdx.x * (blockDim.x >> 5) + wave;
    const int ntiles = Ld / 16;            // 128 query tiles
    const int qt = gw % ntiles;
    const int bh = gw / ntiles;            // 0 .. B*H-1
    if (bh >= Bd * Hd) return;
    const int b = bh / Hd, h = bh % Hd;
    const int q0 = qt * 16;

    __shared__ __bf16 Pbuf[8][16][32];     // per-wave 16x32 probability tile
    __bf16 (*P)[32] = Pbuf[wave];

    // Q fragments for this query tile (dk 0..31, dk 32..63)
    const __bf16* qrow = Q + (size_t)(b * Ld + q0 + ln) * Dd + h * DKd;
    const v16bf qa0 = load_a_frag(qrow, half);
    const v16bf qa1 = load_a_frag(qrow + 32, half);

    v8f o0 = {}, o1 = {}, o2 = {}, o3 = {};
    float mrow[8], lrow[8];
#pragma unroll
    for (int r = 0; r < 8; ++r) { mrow[r] = -3.0e38f; lrow[r] = 0.0f; }

    const __bf16* kbase = Km + (size_t)(b * Ld) * Dd + h * DKd + 16 * half;
    const __bf16* vtb   = Vt + (size_t)bh * DKd * Ld + 16 * half;
    const float scale = 0.125f;            // 1/sqrt(64)
    const int kend = q0 + 16;

    // Prologue: fragments for key block kb = 0.
    v16bf kf00 = load_b_frag(kbase + (size_t)ln * Dd);
    v16bf kf01 = load_b_frag(kbase + (size_t)ln * Dd + 32);
    v16bf kf10 = load_b_frag(kbase + (size_t)(16 + ln) * Dd);
    v16bf kf11 = load_b_frag(kbase + (size_t)(16 + ln) * Dd + 32);
    v16bf vf0  = load_b_frag(vtb + (size_t)(0 * 16 + ln) * Ld);
    v16bf vf1  = load_b_frag(vtb + (size_t)(1 * 16 + ln) * Ld);
    v16bf vf2  = load_b_frag(vtb + (size_t)(2 * 16 + ln) * Ld);
    v16bf vf3  = load_b_frag(vtb + (size_t)(3 * 16 + ln) * Ld);

    for (int kb = 0; kb < kend; kb += 32) {
        // ---- scores S = Q * K^T (two 16x16 tiles, keys kb.. / kb+16..) ----
        v8f s0 = {}, s1 = {};
        s0 = wmma_bf16(qa0, kf00, s0);
        s0 = wmma_bf16(qa1, kf01, s0);
        s1 = wmma_bf16(qa0, kf10, s1);
        s1 = wmma_bf16(qa1, kf11, s1);

        // ---- issue next block's K/V fragment loads (latency hidden under
        //      the softmax below); clamped on the final block ----
        const int kbn = (kb + 32 < kend) ? (kb + 32) : kb;
        const __bf16* kn0 = kbase + (size_t)(kbn + ln) * Dd;
        const __bf16* kn1 = kbase + (size_t)(kbn + 16 + ln) * Dd;
        v16bf nk00 = load_b_frag(kn0);
        v16bf nk01 = load_b_frag(kn0 + 32);
        v16bf nk10 = load_b_frag(kn1);
        v16bf nk11 = load_b_frag(kn1 + 32);
        v16bf nv0  = load_b_frag(vtb + (size_t)(0 * 16 + ln) * Ld + kbn);
        v16bf nv1  = load_b_frag(vtb + (size_t)(1 * 16 + ln) * Ld + kbn);
        v16bf nv2  = load_b_frag(vtb + (size_t)(2 * 16 + ln) * Ld + kbn);
        v16bf nv3  = load_b_frag(vtb + (size_t)(3 * 16 + ln) * Ld + kbn);

        // ---- scale + causal mask + row max (over 16 lanes of each half) ----
        float m_new[8];
#pragma unroll
        for (int r = 0; r < 8; ++r) {
            const int q = q0 + r + 8 * half;
            float v0 = s0[r] * scale;
            float v1 = s1[r] * scale;
            if (kb + ln > q)      v0 = -3.0e38f;
            if (kb + 16 + ln > q) v1 = -3.0e38f;
            s0[r] = v0; s1[r] = v1;
            float mx = fmaxf(v0, v1);
            mx = fmaxf(mx, __shfl_xor(mx, 1, 32));
            mx = fmaxf(mx, __shfl_xor(mx, 2, 32));
            mx = fmaxf(mx, __shfl_xor(mx, 4, 32));
            mx = fmaxf(mx, __shfl_xor(mx, 8, 32));
            m_new[r] = fmaxf(mrow[r], mx);
        }

        // ---- exp, row sums, rescale accumulators, stage P into LDS ----
#pragma unroll
        for (int r = 0; r < 8; ++r) {
            const float e0 = __expf(s0[r] - m_new[r]);
            const float e1 = __expf(s1[r] - m_new[r]);
            float rs = e0 + e1;
            rs += __shfl_xor(rs, 1, 32);
            rs += __shfl_xor(rs, 2, 32);
            rs += __shfl_xor(rs, 4, 32);
            rs += __shfl_xor(rs, 8, 32);
            const float corr = __expf(mrow[r] - m_new[r]);
            lrow[r] = lrow[r] * corr + rs;
            mrow[r] = m_new[r];
            o0[r] *= corr; o1[r] *= corr; o2[r] *= corr; o3[r] *= corr;
            P[r + 8 * half][ln]      = (__bf16)e0;
            P[r + 8 * half][ln + 16] = (__bf16)e1;
        }
        __asm__ volatile("" ::: "memory");   // keep LDS store->load order

        // ---- reload P in A-fragment layout (wave-local LDS transpose) ----
        const v16bf pa = load_a_frag(&P[ln][0], half);
        __asm__ volatile("" ::: "memory");

        // ---- O += P * V  (V fragments preloaded an iteration ahead) ----
        o0 = wmma_bf16(pa, vf0, o0);
        o1 = wmma_bf16(pa, vf1, o1);
        o2 = wmma_bf16(pa, vf2, o2);
        o3 = wmma_bf16(pa, vf3, o3);

        kf00 = nk00; kf01 = nk01; kf10 = nk10; kf11 = nk11;
        vf0 = nv0; vf1 = nv1; vf2 = nv2; vf3 = nv3;
    }

    // ---- normalize and write attention output (bf16, row-major [B*L][D]) ----
#pragma unroll
    for (int r = 0; r < 8; ++r) {
        const float inv = 1.0f / lrow[r];
        const size_t base = (size_t)(b * Ld + q0 + r + 8 * half) * Dd + h * DKd;
        AO[base + 0 * 16 + ln] = (__bf16)(o0[r] * inv);
        AO[base + 1 * 16 + ln] = (__bf16)(o1[r] * inv);
        AO[base + 2 * 16 + ln] = (__bf16)(o2[r] * inv);
        AO[base + 3 * 16 + ln] = (__bf16)(o3[r] * inv);
    }
}

// ---------------------------------------------------------------------------
// Launch
// ---------------------------------------------------------------------------
extern "C" void kernel_launch(void* const* d_in, const int* in_sizes, int n_in,
                              void* d_out, int out_size, void* d_ws, size_t ws_size,
                              hipStream_t stream) {
    const float* Xc = (const float*)d_in[0];
    const float* Xv = (const float*)d_in[1];
    // d_in[2] = mask (causal, implicit in kernel)
    const float* Wq = (const float*)d_in[3];
    const float* bq = (const float*)d_in[4];
    const float* Wk = (const float*)d_in[5];
    const float* bk = (const float*)d_in[6];
    const float* Wv = (const float*)d_in[7];
    const float* bv = (const float*)d_in[8];
    const float* Wo = (const float*)d_in[9];
    const float* bo = (const float*)d_in[10];
    float* out = (float*)d_out;

    char* ws = (char*)d_ws;
    size_t off = 0;
    auto alloc = [&](size_t bytes) -> void* {
        void* p = ws + off;
        off += (bytes + 255) & ~(size_t)255;
        return p;
    };
    const size_t actBytes = (size_t)BLd * Dd * sizeof(__bf16);
    const size_t wBytes   = (size_t)Dd * Dd * sizeof(__bf16);
    __bf16* xc  = (__bf16*)alloc(actBytes);
    __bf16* xv  = (__bf16*)alloc(actBytes);
    __bf16* wqt = (__bf16*)alloc(wBytes);
    __bf16* wkt = (__bf16*)alloc(wBytes);
    __bf16* wvt = (__bf16*)alloc(wBytes);
    __bf16* wot = (__bf16*)alloc(wBytes);
    __bf16* qb  = (__bf16*)alloc(actBytes);
    __bf16* kbuf= (__bf16*)alloc(actBytes);
    __bf16* vt  = (__bf16*)alloc(actBytes);   // [B*H][DK][L]
    __bf16* ao  = (__bf16*)alloc(actBytes);
    (void)ws_size; (void)in_sizes; (void)n_in; (void)out_size;

    // 1) conversions
    cvt_f32_bf16_kernel<<<2048, 256, 0, stream>>>(Xc, xc, (size_t)BLd * Dd);
    cvt_f32_bf16_kernel<<<2048, 256, 0, stream>>>(Xv, xv, (size_t)BLd * Dd);
    transpose_w_kernel<<<1024, 256, 0, stream>>>(Wq, wqt, Dd);
    transpose_w_kernel<<<1024, 256, 0, stream>>>(Wk, wkt, Dd);
    transpose_w_kernel<<<1024, 256, 0, stream>>>(Wv, wvt, Dd);
    transpose_w_kernel<<<1024, 256, 0, stream>>>(Wo, wot, Dd);

    // 2) QKV projections (WMMA).  waves = (M/16)*(N/64) = 4096 -> 512 blocks
    gemm_bf16_kernel<<<512, 256, 0, stream>>>(xc, wqt, bq, qb,   BLd, Dd, Dd, 0);
    gemm_bf16_kernel<<<512, 256, 0, stream>>>(xc, wkt, bk, kbuf, BLd, Dd, Dd, 0);
    gemm_bf16_kernel<<<512, 256, 0, stream>>>(xv, wvt, bv, vt,   BLd, Dd, Dd, 1);

    // 3) flash attention: B*H*(L/16) = 4096 waves -> 512 blocks
    flash_attn_kernel<<<512, 256, 0, stream>>>(qb, kbuf, vt, ao);

    // 4) output projection (fp32 out)
    gemm_bf16_kernel<<<512, 256, 0, stream>>>(ao, wot, bo, out, BLd, Dd, Dd, 2);
}
